// Encoder_63471026700290
// MI455X (gfx1250) — matrix operand order repaired and checked by reading, not compiled, not run on hardware
//
#include <hip/hip_runtime.h>

// ---------------- model constants ----------------
#define LNUM   6
#define HID    768
#define NHEAD  12
#define HDIM   64
#define INTERS 2048
#define SEQ    512
#define NB     8
#define EPSLN  1e-7f
#define ATT_SCALE 0.07216878364870322f   // 1/sqrt(3*HD) = 1/sqrt(192)

typedef __attribute__((ext_vector_type(16))) __bf16 v16bf;
typedef __attribute__((ext_vector_type(8)))  __bf16 bf16x8;
typedef __attribute__((ext_vector_type(8)))  float  v8f;

// D = A(16x32 bf16) * B(32x16 bf16) + C (f32)
__device__ __forceinline__ v8f wmma_bf16(v16bf a, v16bf b, v8f c) {
  return __builtin_amdgcn_wmma_f32_16x16x32_bf16(false, a, false, b, (short)0, c,
                                                 false, false);
}

// A fragment: lane holds one row; halves[0..7]=K(hi*8..), halves[8..15]=K(16+hi*8..)
__device__ __forceinline__ v16bf frag_a(const __bf16* r, int hi) {
  bf16x8 lo = *(const bf16x8*)(r + hi * 8);
  bf16x8 hh = *(const bf16x8*)(r + 16 + hi * 8);
  return __builtin_shufflevector(lo, hh, 0,1,2,3,4,5,6,7,8,9,10,11,12,13,14,15);
}
// B fragment: lane holds one column (a row of B^T); halves[0..7]=K(hi*8..), halves[8..15]=K(16+hi*8..)
__device__ __forceinline__ v16bf frag_b(const __bf16* r, int hi) {
  bf16x8 lo = *(const bf16x8*)(r + hi * 8);
  bf16x8 hh = *(const bf16x8*)(r + 16 + hi * 8);
  return __builtin_shufflevector(lo, hh, 0,1,2,3,4,5,6,7,8,9,10,11,12,13,14,15);
}

// ---------------- generic WMMA GEMM: C = A[M,K] * Bw[N,K]^T (+bias) ----------------
// 128 threads = 4 waves (2x2); macro tile 128x128, K step 32; wave tile 64x64
// (16 WMMA per 8 LDS fragment loads). Double-buffered LDS, register-staged
// global prefetch, one barrier per K-step, global_prefetch_b8 two steps ahead.
// GUARD=false for exact-tile GEMMs (no bounds checks in loads or epilogue).
#define BM  128
#define BN  128
#define BKK 32
#define LDT 40   // padded LDS row stride (halves)

template <int MODE, bool GUARD>   // MODE 0: f32 store, 1: bf16 store, 2: f32 +=
__global__ void __launch_bounds__(128)
gemm_kernel(const __bf16* __restrict__ A,  int lda, int aSB, int aSH,
            const __bf16* __restrict__ Bw, int ldb, int bSB, int bSH,
            void* __restrict__ Cv,         int ldc, int cSB, int cSH,
            const float* __restrict__ bias, int M, int N, int K, int nh) {
  __shared__ __attribute__((aligned(16))) __bf16 As[2][BM][LDT];
  __shared__ __attribute__((aligned(16))) __bf16 Bs[2][BN][LDT];

  int z  = blockIdx.z;
  int zb = z / nh, zh = z - zb * nh;
  const __bf16* Ab = A  + zb * aSB + zh * aSH;
  const __bf16* Bb = Bw + zb * bSB + zh * bSH;
  int cOff = zb * cSB + zh * cSH;

  int tid  = threadIdx.x;
  int wave = tid >> 5, lane = tid & 31;
  int l15 = lane & 15, hi = lane >> 4;
  int wm = (wave >> 1) & 1, wn = wave & 1;
  int tile_m = blockIdx.y * BM, tile_n = blockIdx.x * BN;

  v8f vz = {0.f,0.f,0.f,0.f,0.f,0.f,0.f,0.f};
  v8f acc[4][4];
#pragma unroll
  for (int i = 0; i < 4; ++i)
#pragma unroll
    for (int j = 0; j < 4; ++j) acc[i][j] = vz;

  __bf16 bz = (__bf16)0.0f;
  bf16x8 z8 = {bz,bz,bz,bz,bz,bz,bz,bz};

  // one row per thread (tid 0..127)
  int gr = tile_m + tid;
  int gn = tile_n + tid;
  bool am = GUARD ? (gr < M) : true;
  bool bm = GUARD ? (gn < N) : true;
  const __bf16* aRow = Ab + gr * lda;
  const __bf16* bRow = Bb + gn * ldb;

  bf16x8 ar[4], br[4];
  // load first K-step into registers
#pragma unroll
  for (int u = 0; u < 4; ++u) {
    ar[u] = am ? *(const bf16x8*)(aRow + u * 8) : z8;
    br[u] = bm ? *(const bf16x8*)(bRow + u * 8) : z8;
  }

  int buf = 0;
  for (int k0 = 0; k0 < K; k0 += BKK) {
    // stage current registers to LDS
#pragma unroll
    for (int u = 0; u < 4; ++u) {
      *(bf16x8*)&As[buf][tid][u * 8] = ar[u];
      *(bf16x8*)&Bs[buf][tid][u * 8] = br[u];
    }
    __syncthreads();

    // prefetch next K-step into registers (overlaps with WMMA below)
    int kn = k0 + BKK;
    if (kn < K) {
#pragma unroll
      for (int u = 0; u < 4; ++u) {
        ar[u] = am ? *(const bf16x8*)(aRow + kn + u * 8) : z8;
        br[u] = bm ? *(const bf16x8*)(bRow + kn + u * 8) : z8;
      }
      if (kn + BKK < K) {   // warm L2/WGP$ two steps ahead (global_prefetch_b8)
        if (am) __builtin_prefetch(aRow + kn + BKK, 0, 1);
        if (bm) __builtin_prefetch(bRow + kn + BKK, 0, 1);
      }
    }

    v16bf bfr[4];
#pragma unroll
    for (int j = 0; j < 4; ++j)
      bfr[j] = frag_b(&Bs[buf][wn * 64 + j * 16 + l15][0], hi);
#pragma unroll
    for (int i = 0; i < 4; ++i) {
      v16bf af = frag_a(&As[buf][wm * 64 + i * 16 + l15][0], hi);
#pragma unroll
      for (int j = 0; j < 4; ++j) acc[i][j] = wmma_bf16(af, bfr[j], acc[i][j]);
    }
    buf ^= 1;
  }

  float*  Cf = (float*)Cv;
  __bf16* Cb = (__bf16*)Cv;
#pragma unroll
  for (int j = 0; j < 4; ++j) {
    int gnn = tile_n + wn * 64 + j * 16 + l15;
    float bv_ = (bias && (!GUARD || gnn < N)) ? bias[gnn] : 0.f;
#pragma unroll
    for (int i = 0; i < 4; ++i)
#pragma unroll
      for (int v = 0; v < 8; ++v) {
        int gm = tile_m + wm * 64 + i * 16 + hi * 8 + v;
        if (!GUARD || (gm < M && gnn < N)) {
          float val = acc[i][j][v] + bv_;
          int o = cOff + gm * ldc + gnn;
          if (MODE == 0)      Cf[o] = val;
          else if (MODE == 1) Cb[o] = (__bf16)val;
          else                Cf[o] += val;
        }
      }
  }
}

// ---------------- fused flash attention with disentangled rel-pos ----------------
// grid: (SEQ/64, NB*NHEAD), block 128 (4 waves); wave owns 16 q rows.
__global__ void __launch_bounds__(128)
attn_kernel(const __bf16* __restrict__ Q, const __bf16* __restrict__ Kk,
            const __bf16* __restrict__ VT,
            const float* __restrict__ CP, const float* __restrict__ PC,
            const int* __restrict__ IDX, const unsigned char* __restrict__ MSK,
            __bf16* __restrict__ OUT) {
  __shared__ __attribute__((aligned(16))) __bf16 Ps[4][16][72];
  int z = blockIdx.y;
  int b = z / NHEAD, h = z - b * NHEAD;
  int wave = threadIdx.x >> 5, lane = threadIdx.x & 31;
  int l15 = lane & 15, hi = lane >> 4;
  int qbase = blockIdx.x * 64 + wave * 16;
  const float NEG_INF = -__builtin_inff();

  const __bf16* qr = Q + ((qbase + l15) * NB + b) * HID + h * HDIM;
  v16bf qf0 = frag_a(qr, hi);
  v16bf qf1 = frag_a(qr + 32, hi);

  v8f vz = {0.f,0.f,0.f,0.f,0.f,0.f,0.f,0.f};
  v8f O[4] = {vz, vz, vz, vz};
  float mi[8], li[8];
#pragma unroll
  for (int v = 0; v < 8; ++v) { mi[v] = -1e30f; li[v] = 0.f; }

  for (int kt = 0; kt < SEQ; kt += 64) {
    // warm next K/V tile while this one computes
    if (kt + 64 < SEQ) {
      int kr2 = kt + 64 + lane;
      __builtin_prefetch(Kk + (kr2 * NB + b) * HID + h * HDIM, 0, 1);
      __builtin_prefetch(Kk + ((kr2 + 32) * NB + b) * HID + h * HDIM, 0, 1);
      __builtin_prefetch(VT + (z * HDIM + lane) * SEQ + kt + 64, 0, 1);
      __builtin_prefetch(VT + (z * HDIM + lane + 32) * SEQ + kt + 64, 0, 1);
    }
    v8f sa[4] = {vz, vz, vz, vz};
#pragma unroll
    for (int nt = 0; nt < 4; ++nt) {
      const __bf16* kr = Kk + ((kt + nt * 16 + l15) * NB + b) * HID + h * HDIM;
      sa[nt] = wmma_bf16(qf0, frag_b(kr, hi),      sa[nt]);
      sa[nt] = wmma_bf16(qf1, frag_b(kr + 32, hi), sa[nt]);
    }
    // scale + rel-pos gathers + mask (element mapping: m=hi*8+v, n=l15)
    float sc[4][8];
#pragma unroll
    for (int nt = 0; nt < 4; ++nt) {
      int kg = kt + nt * 16 + l15;
      bool m = MSK[b * SEQ + kg] != 0;
#pragma unroll
      for (int v = 0; v < 8; ++v) {
        int qg = qbase + hi * 8 + v;
        int j  = IDX[qg * SEQ + kg];
        float val = (sa[nt][v] + CP[(z * SEQ + qg) * 64 + j]
                               + PC[(z * SEQ + kg) * 64 + j]) * ATT_SCALE;
        sc[nt][v] = m ? NEG_INF : val;
      }
    }
    // online softmax (row spread over 16 lanes of a half-wave)
#pragma unroll
    for (int v = 0; v < 8; ++v) {
      float mx = fmaxf(fmaxf(sc[0][v], sc[1][v]), fmaxf(sc[2][v], sc[3][v]));
#pragma unroll
      for (int off = 1; off < 16; off <<= 1) mx = fmaxf(mx, __shfl_xor(mx, off, 32));
      float mnew = fmaxf(mi[v], mx);
      float f = __expf(mi[v] - mnew);
      float rs = 0.f;
#pragma unroll
      for (int nt = 0; nt < 4; ++nt) {
        float p = __expf(sc[nt][v] - mnew);
        rs += p;
        Ps[wave][hi * 8 + v][nt * 16 + l15] = (__bf16)p;
      }
#pragma unroll
      for (int off = 1; off < 16; off <<= 1) rs += __shfl_xor(rs, off, 32);
      li[v] = li[v] * f + rs;
      mi[v] = mnew;
#pragma unroll
      for (int dt = 0; dt < 4; ++dt) O[dt][v] *= f;
    }
    // CDNA5 split wait: make per-wave LDS P-tile visible before re-reading as A frags
    asm volatile("s_wait_dscnt 0x0" ::: "memory");
#pragma unroll
    for (int ks = 0; ks < 2; ++ks) {
      v16bf pf = frag_a(&Ps[wave][l15][ks * 32], hi);
#pragma unroll
      for (int dt = 0; dt < 4; ++dt) {
        const __bf16* vr = VT + (z * HDIM + dt * 16 + l15) * SEQ + kt + ks * 32;
        O[dt] = wmma_bf16(pf, frag_b(vr, hi), O[dt]);
      }
    }
  }
#pragma unroll
  for (int dt = 0; dt < 4; ++dt)
#pragma unroll
    for (int v = 0; v < 8; ++v) {
      int qg = qbase + hi * 8 + v;
      float val = (li[v] > 0.f) ? O[dt][v] / li[v] : 0.f;
      OUT[(qg * NB + b) * HID + h * HDIM + dt * 16 + l15] = (__bf16)val;
    }
}

// ---------------- LayerNorm family (one 256-thread block per row) ----------------
__global__ void __launch_bounds__(256)
ln_kernel(const float* __restrict__ in, const float* __restrict__ g,
          const float* __restrict__ bb, float* __restrict__ out_f,
          __bf16* __restrict__ out_b) {
  int row = blockIdx.x;
  const float* x = in + row * HID;
  float v0[3], s = 0.f, s2 = 0.f;
#pragma unroll
  for (int i = 0; i < 3; ++i) {
    float t = x[threadIdx.x + i * 256];
    v0[i] = t; s += t; s2 += t * t;
  }
#pragma unroll
  for (int off = 16; off > 0; off >>= 1) {
    s += __shfl_xor(s, off, 32); s2 += __shfl_xor(s2, off, 32);
  }
  __shared__ float ra[8], rb[8];
  int w = threadIdx.x >> 5;
  if ((threadIdx.x & 31) == 0) { ra[w] = s; rb[w] = s2; }
  __syncthreads();
  float ts = 0.f, ts2 = 0.f;
#pragma unroll
  for (int i = 0; i < 8; ++i) { ts += ra[i]; ts2 += rb[i]; }
  float mean = ts * (1.f / HID);
  float var  = ts2 * (1.f / HID) - mean * mean;
  float r = rsqrtf(var + EPSLN);
#pragma unroll
  for (int i = 0; i < 3; ++i) {
    int c = threadIdx.x + i * 256;
    float y = (v0[i] - mean) * r;
    if (g) y = y * g[c] + bb[c];
    if (out_f) out_f[row * HID + c] = y;
    if (out_b) out_b[row * HID + c] = (__bf16)y;
  }
}

__global__ void __launch_bounds__(256)
ln_residual_kernel(const float* __restrict__ y, const float* __restrict__ g,
                   const float* __restrict__ bb, float* __restrict__ x) {
  int row = blockIdx.x;
  const float* yy = y + row * HID;
  float v0[3], s = 0.f, s2 = 0.f;
#pragma unroll
  for (int i = 0; i < 3; ++i) {
    float t = yy[threadIdx.x + i * 256];
    v0[i] = t; s += t; s2 += t * t;
  }
#pragma unroll
  for (int off = 16; off > 0; off >>= 1) {
    s += __shfl_xor(s, off, 32); s2 += __shfl_xor(s2, off, 32);
  }
  __shared__ float ra[8], rb[8];
  int w = threadIdx.x >> 5;
  if ((threadIdx.x & 31) == 0) { ra[w] = s; rb[w] = s2; }
  __syncthreads();
  float ts = 0.f, ts2 = 0.f;
#pragma unroll
  for (int i = 0; i < 8; ++i) { ts += ra[i]; ts2 += rb[i]; }
  float mean = ts * (1.f / HID);
  float var  = ts2 * (1.f / HID) - mean * mean;
  float r = rsqrtf(var + EPSLN);
#pragma unroll
  for (int i = 0; i < 3; ++i) {
    int c = threadIdx.x + i * 256;
    x[row * HID + c] += (v0[i] - mean) * r * g[c] + bb[c];
  }
}

__global__ void __launch_bounds__(256)
geglu_ln_kernel(const float* __restrict__ hh, __bf16* __restrict__ out) {
  int row = blockIdx.x;
  const float* a  = hh + row * (2 * INTERS);
  const float* gt = a + INTERS;
  float v0[8], s = 0.f, s2 = 0.f;
#pragma unroll
  for (int i = 0; i < 8; ++i) {
    int c = threadIdx.x + i * 256;
    float xg = gt[c];
    float u  = 0.7978845608028654f * (xg + 0.044715f * xg * xg * xg);
    float t  = a[c] * 0.5f * xg * (1.f + tanhf(u));
    v0[i] = t; s += t; s2 += t * t;
  }
#pragma unroll
  for (int off = 16; off > 0; off >>= 1) {
    s += __shfl_xor(s, off, 32); s2 += __shfl_xor(s2, off, 32);
  }
  __shared__ float ra[8], rb[8];
  int w = threadIdx.x >> 5;
  if ((threadIdx.x & 31) == 0) { ra[w] = s; rb[w] = s2; }
  __syncthreads();
  float ts = 0.f, ts2 = 0.f;
#pragma unroll
  for (int i = 0; i < 8; ++i) { ts += ra[i]; ts2 += rb[i]; }
  float mean = ts * (1.f / INTERS);
  float var  = ts2 * (1.f / INTERS) - mean * mean;
  float r = rsqrtf(var + EPSLN);
#pragma unroll
  for (int i = 0; i < 8; ++i) {
    int c = threadIdx.x + i * 256;
    out[row * INTERS + c] = (__bf16)((v0[i] - mean) * r);
  }
}

// ---------------- small utility kernels ----------------
__global__ void cvt_kernel(const float* __restrict__ in, __bf16* __restrict__ out,
                           int n) {
  int i = blockIdx.x * blockDim.x + threadIdx.x;
  if (i < n) out[i] = (__bf16)in[i];
}

// v[(s*NB+b)*HID + h*HDIM + d] -> vT[((b*NHEAD+h)*HDIM + d)*SEQ + s]
__global__ void vtrans_kernel(const __bf16* __restrict__ v,
                              __bf16* __restrict__ vt) {
  int i = blockIdx.x * blockDim.x + threadIdx.x;
  if (i >= SEQ * NB * HID) return;
  int d = i & 63;
  int t = i >> 6;
  int h = t % NHEAD; t /= NHEAD;
  int b = t % NB;
  int sidx = t / NB;
  vt[((b * NHEAD + h) * HDIM + d) * SEQ + sidx] = v[i];
}

// ---------------- host orchestration ----------------
extern "C" void kernel_launch(void* const* d_in, const int* in_sizes, int n_in,
                              void* d_out, int out_size, void* d_ws, size_t ws_size,
                              hipStream_t stream) {
  (void)in_sizes; (void)n_in; (void)out_size;
  const float* hid0    = (const float*)d_in[0];
  const unsigned char* msk = (const unsigned char*)d_in[1];
  const int*   pidx    = (const int*)d_in[2];
  const float* rel_emb = (const float*)d_in[3];
  const float* rel_g   = (const float*)d_in[4];
  const float* rel_bv  = (const float*)d_in[5];
  const float* Wq = (const float*)d_in[6];  const float* bq = (const float*)d_in[7];
  const float* Wk = (const float*)d_in[8];  const float* bk = (const float*)d_in[9];
  const float* Wv = (const float*)d_in[10]; const float* bv = (const float*)d_in[11];
  const float* Wo = (const float*)d_in[12]; const float* bo = (const float*)d_in[13];
  const float* pg = (const float*)d_in[14]; const float* pb = (const float*)d_in[15];
  const float* W1 = (const float*)d_in[16]; const float* W2 = (const float*)d_in[17];
  float* x = (float*)d_out;

  const int ROWS = SEQ * NB;  // 4096
  char* base = (char*)d_ws;
  size_t off = 0;
  auto alloc = [&](size_t bytes) -> void* {
    void* p = base + off;
    off = (off + bytes + 255) & ~(size_t)255;
    return p;
  };
  __bf16* wq_bf = (__bf16*)alloc((size_t)LNUM * HID * HID * 2);
  __bf16* wk_bf = (__bf16*)alloc((size_t)LNUM * HID * HID * 2);
  __bf16* wv_bf = (__bf16*)alloc((size_t)LNUM * HID * HID * 2);
  __bf16* wo_bf = (__bf16*)alloc((size_t)LNUM * HID * HID * 2);
  __bf16* w1_bf = (__bf16*)alloc((size_t)LNUM * 2 * INTERS * HID * 2);
  __bf16* w2_bf = (__bf16*)alloc((size_t)LNUM * HID * INTERS * 2);
  __bf16* rel_bf  = (__bf16*)alloc((size_t)64 * HID * 2);
  __bf16* qpos_bf = (__bf16*)alloc((size_t)64 * HID * 2);
  __bf16* kpos_bf = (__bf16*)alloc((size_t)64 * HID * 2);
  __bf16* xn_bf  = (__bf16*)alloc((size_t)ROWS * HID * 2);
  __bf16* q_bf   = (__bf16*)alloc((size_t)ROWS * HID * 2);
  __bf16* k_bf   = (__bf16*)alloc((size_t)ROWS * HID * 2);
  __bf16* v_bf   = (__bf16*)alloc((size_t)ROWS * HID * 2);
  __bf16* ctx_bf = (__bf16*)alloc((size_t)ROWS * HID * 2);
  __bf16* vt_bf  = (__bf16*)alloc((size_t)NB * NHEAD * HDIM * SEQ * 2);
  __bf16* t_bf   = (__bf16*)alloc((size_t)ROWS * INTERS * 2);
  float*  cp_f   = (float*)alloc((size_t)NB * NHEAD * SEQ * 64 * 4);
  float*  pc_f   = (float*)alloc((size_t)NB * NHEAD * SEQ * 64 * 4);
  float*  y_f    = (float*)alloc((size_t)ROWS * HID * 4);
  float*  h_f    = (float*)alloc((size_t)ROWS * 2 * INTERS * 4);
  if (off > ws_size) return;  // workspace too small: refuse to scribble

  hipMemcpyAsync(d_out, hid0, (size_t)ROWS * HID * 4, hipMemcpyDeviceToDevice, stream);

  auto cvt = [&](const float* in, __bf16* out, int n) {
    cvt_kernel<<<dim3((unsigned)((n + 255) / 256)), dim3(256), 0, stream>>>(in, out, n);
  };
  cvt(Wq, wq_bf, LNUM * HID * HID);
  cvt(Wk, wk_bf, LNUM * HID * HID);
  cvt(Wv, wv_bf, LNUM * HID * HID);
  cvt(Wo, wo_bf, LNUM * HID * HID);
  cvt(W1, w1_bf, LNUM * 2 * INTERS * HID);
  cvt(W2, w2_bf, LNUM * HID * INTERS);

  // rel = LN(rel_emb, g, b) -> bf16  (63 rows)
  ln_kernel<<<63, 256, 0, stream>>>(rel_emb, rel_g, rel_bv, nullptr, rel_bf);

  dim3 gProj((HID + BN - 1) / BN, (ROWS + BM - 1) / BM, 1);
  dim3 gPos((HID + BN - 1) / BN, 1, 1);
  dim3 gHead(1, (SEQ + BM - 1) / BM, NB * NHEAD);
  dim3 gFF1((2 * INTERS + BN - 1) / BN, (ROWS + BM - 1) / BM, 1);
  dim3 gFF2((HID + BN - 1) / BN, (ROWS + BM - 1) / BM, 1);

  for (int l = 0; l < LNUM; ++l) {
    const __bf16* wql = wq_bf + (size_t)l * HID * HID;
    const __bf16* wkl = wk_bf + (size_t)l * HID * HID;
    const __bf16* wvl = wv_bf + (size_t)l * HID * HID;
    const __bf16* wol = wo_bf + (size_t)l * HID * HID;
    const __bf16* w1l = w1_bf + (size_t)l * 2 * INTERS * HID;
    const __bf16* w2l = w2_bf + (size_t)l * HID * INTERS;
    const float* bql = bq + (size_t)l * HID;
    const float* bkl = bk + (size_t)l * HID;
    const float* bvl = bv + (size_t)l * HID;
    const float* bol = bo + (size_t)l * HID;

    // xn = LN(x) (no affine), bf16
    ln_kernel<<<ROWS, 256, 0, stream>>>(x, nullptr, nullptr, nullptr, xn_bf);

    // q/k/v projections (bf16 out, +bias) — exact tiles, unguarded
    gemm_kernel<1,false><<<gProj, 128, 0, stream>>>(xn_bf, HID, 0, 0, wql, HID, 0, 0,
                                                    q_bf, HID, 0, 0, bql, ROWS, HID, HID, 1);
    gemm_kernel<1,false><<<gProj, 128, 0, stream>>>(xn_bf, HID, 0, 0, wkl, HID, 0, 0,
                                                    k_bf, HID, 0, 0, bkl, ROWS, HID, HID, 1);
    gemm_kernel<1,false><<<gProj, 128, 0, stream>>>(xn_bf, HID, 0, 0, wvl, HID, 0, 0,
                                                    v_bf, HID, 0, 0, bvl, ROWS, HID, HID, 1);
    vtrans_kernel<<<(ROWS * HID + 255) / 256, 256, 0, stream>>>(v_bf, vt_bf);

    // rel-pos projections (63x768) — M ragged, guarded
    gemm_kernel<1,true><<<gPos, 128, 0, stream>>>(rel_bf, HID, 0, 0, wql, HID, 0, 0,
                                                  qpos_bf, HID, 0, 0, bql, 63, HID, HID, 1);
    gemm_kernel<1,true><<<gPos, 128, 0, stream>>>(rel_bf, HID, 0, 0, wkl, HID, 0, 0,
                                                  kpos_bf, HID, 0, 0, bkl, 63, HID, HID, 1);

    // cp[b,h,q,j] = q_head . kpos_head ; pcT[b,h,k,j] = k_head . qpos_head — N ragged
    gemm_kernel<0,true><<<gHead, 128, 0, stream>>>(q_bf, NB * HID, HID, HDIM,
                                                   kpos_bf, HID, 0, HDIM,
                                                   cp_f, 64, NHEAD * SEQ * 64, SEQ * 64,
                                                   nullptr, SEQ, 63, HDIM, NHEAD);
    gemm_kernel<0,true><<<gHead, 128, 0, stream>>>(k_bf, NB * HID, HID, HDIM,
                                                   qpos_bf, HID, 0, HDIM,
                                                   pc_f, 64, NHEAD * SEQ * 64, SEQ * 64,
                                                   nullptr, SEQ, 63, HDIM, NHEAD);

    // fused attention
    attn_kernel<<<dim3(SEQ / 64, NB * NHEAD), 128, 0, stream>>>(
        q_bf, k_bf, vt_bf, cp_f, pc_f, pidx, msk, ctx_bf);

    // out projection (f32) then x += LN(y, pg, pb)
    gemm_kernel<0,false><<<gProj, 128, 0, stream>>>(ctx_bf, HID, 0, 0, wol, HID, 0, 0,
                                                    y_f, HID, 0, 0, bol, ROWS, HID, HID, 1);
    ln_residual_kernel<<<ROWS, 256, 0, stream>>>(
        y_f, pg + (size_t)l * HID, pb + (size_t)l * HID, x);

    // FFN: h = LN(x) @ W1^T ; t = LN(GeGLU(h)) ; x += t @ W2^T
    ln_kernel<<<ROWS, 256, 0, stream>>>(x, nullptr, nullptr, nullptr, xn_bf);
    gemm_kernel<0,false><<<gFF1, 128, 0, stream>>>(xn_bf, HID, 0, 0, w1l, HID, 0, 0,
                                                   h_f, 2 * INTERS, 0, 0, nullptr,
                                                   ROWS, 2 * INTERS, HID, 1);
    geglu_ln_kernel<<<ROWS, 256, 0, stream>>>(h_f, t_bf);
    gemm_kernel<2,false><<<gFF2, 128, 0, stream>>>(t_bf, INTERS, 0, 0, w2l, INTERS, 0, 0,
                                                   x, HID, 0, 0, nullptr,
                                                   ROWS, HID, INTERS, 1);
  }
}
